// Visual_Modality_Attention_48850958024952
// MI455X (gfx1250) — matrix operand order repaired and checked by reading, not compiled, not run on hardware
//
#include <hip/hip_runtime.h>
#include <hip/hip_bf16.h>
#include <stdint.h>

// ---------------------------------------------------------------------------
// MI455X / gfx1250 implementation of Visual_Modality_Attention.
//
// Precision strategy: all GEMM-shaped work runs on v_wmma_f32_16x16x32_bf16
// (bf16 inputs, f32 accumulate). Softmax runs in f32 on materialized score
// matrices (B*S*S f32 = 67 MB, L2-resident on the 192 MB L2). The dominant
// bilinear contraction (275 of ~330 GFLOP) keeps one 256x256 bf16 W[o] slice
// (128 KB) resident in LDS per workgroup (staged by the Tensor Data Mover
// where available) and streams token tiles through it.
// ---------------------------------------------------------------------------

typedef __attribute__((ext_vector_type(16))) __bf16        v16bf;
typedef __attribute__((ext_vector_type(8)))  float         v8f;
typedef __attribute__((ext_vector_type(4)))  unsigned int  u32x4;
typedef __attribute__((ext_vector_type(8)))  int           i32x8;
typedef __attribute__((ext_vector_type(4)))  int           i32x4;

#if defined(__HIP_DEVICE_COMPILE__) && __has_builtin(__builtin_amdgcn_tensor_load_to_lds) && __has_builtin(__builtin_amdgcn_s_wait_tensorcnt)
#define VMA_USE_TDM 1
#else
#define VMA_USE_TDM 0
#endif

union Frag16 { v16bf v; u32x4 q[2]; };   // 32 bytes: one WMMA A or B fragment

__device__ __forceinline__ unsigned short f2bf(float x) {
  union { float f; unsigned int u; } v; v.f = x;
  unsigned int r = v.u + 0x7FFFu + ((v.u >> 16) & 1u);   // round-nearest-even
  return (unsigned short)(r >> 16);
}
__device__ __forceinline__ float bf2f(unsigned short h) {
  union { unsigned int u; float f; } v; v.u = ((unsigned int)h) << 16; return v.f;
}
__device__ __forceinline__ v8f wmma_bf16(v16bf a, v16bf b, v8f c) {
  // (neg_a, A, neg_b, B, c_mod, C, reuse_a, reuse_b)
  return __builtin_amdgcn_wmma_f32_16x16x32_bf16(false, a, false, b, (short)0, c,
                                                 false, false);
}
// A frag (16x32 bf16): row = lane%16, k-chunks at half*8 and 16+half*8.
__device__ __forceinline__ void load_a(Frag16& a, const unsigned short* p) {
  a.q[0] = *(const u32x4*)(p);
  a.q[1] = *(const u32x4*)(p + 16);
}
// B frag (32x16 bf16): col = lane%16, k-chunks at half*8 and 16+half*8.
__device__ __forceinline__ void load_b(Frag16& b, const unsigned short* p) {
  b.q[0] = *(const u32x4*)(p);
  b.q[1] = *(const u32x4*)(p + 16);
}

// ---------------------------------------------------------------------------
// f32 -> bf16 elementwise cast
// ---------------------------------------------------------------------------
__global__ void __launch_bounds__(256)
cvt_f32_bf16(const float* __restrict__ in, unsigned short* __restrict__ out,
             long long n) {
  long long i = (long long)blockIdx.x * 256 + threadIdx.x;
  long long stride = (long long)gridDim.x * 256;
  for (; i < n; i += stride) out[i] = f2bf(in[i]);
}

// ---------------------------------------------------------------------------
// Batched bf16 transpose: in [rows x cols] -> out [cols x rows], per blockIdx.z
// ---------------------------------------------------------------------------
__global__ void __launch_bounds__(256)
transpose_bf16(const unsigned short* __restrict__ in,
               unsigned short* __restrict__ out, int rows, int cols) {
  __shared__ unsigned short tile[32][33];
  const size_t bo = (size_t)blockIdx.z * rows * cols;
  const int r0 = blockIdx.y * 32, c0 = blockIdx.x * 32;
  const int tx = threadIdx.x & 31, ty = threadIdx.x >> 5;
  for (int i = ty; i < 32; i += 8)
    tile[i][tx] = in[bo + (size_t)(r0 + i) * cols + (c0 + tx)];
  __syncthreads();
  for (int i = ty; i < 32; i += 8)
    out[bo + (size_t)(c0 + i) * rows + (r0 + tx)] = tile[tx][i];
}

// ---------------------------------------------------------------------------
// Row softmax (f32 in) * scale -> bf16 out. One 256-thread block per row.
// ---------------------------------------------------------------------------
__global__ void __launch_bounds__(256)
softmax_scale_bf16(const float* __restrict__ S, unsigned short* __restrict__ P,
                   int cols, float scale) {
  __shared__ float red[256];
  const size_t row = blockIdx.x;
  const float* s = S + row * cols;
  unsigned short* p = P + row * cols;
  const int tid = threadIdx.x;
  float lmax = -3.402823466e38f;
  for (int c = tid; c < cols; c += 256) lmax = fmaxf(lmax, s[c]);
  red[tid] = lmax; __syncthreads();
  for (int o = 128; o > 0; o >>= 1) {
    if (tid < o) red[tid] = fmaxf(red[tid], red[tid + o]);
    __syncthreads();
  }
  const float m = red[0];
  __syncthreads();
  float lsum = 0.f;
  for (int c = tid; c < cols; c += 256) lsum += __expf(s[c] - m);
  red[tid] = lsum; __syncthreads();
  for (int o = 128; o > 0; o >>= 1) {
    if (tid < o) red[tid] += red[tid + o];
    __syncthreads();
  }
  const float inv = scale / red[0];
  for (int c = tid; c < cols; c += 256) p[c] = f2bf(__expf(s[c] - m) * inv);
}

// ---------------------------------------------------------------------------
// Generic batched "NT" WMMA GEMM: C[m,n] = alpha * sum_k A[m,k]*B[n,k] + bias[n]
//                                          (+ residual[m,n])
// A: [M x K] bf16 row-major, B: [N x K] bf16 row-major. Output f32 or bf16.
// 256 threads = 8 waves; each wave owns a 16(M) x 64(N) strip: one A fragment
// feeds 4 WMMAs. The k-loop is double-buffered in steps of 64 (K % 64 == 0
// for every GEMM in this pipeline) so global_load_b128s for the next step are
// in flight while the current step's WMMAs execute.
// ---------------------------------------------------------------------------
__global__ void __launch_bounds__(256)
gemm_nt_wmma(const unsigned short* __restrict__ A,
             const unsigned short* __restrict__ B,
             const float* __restrict__ bias,
             float* __restrict__ outF, unsigned short* __restrict__ outH,
             const float* __restrict__ alphaPtr,
             const float* __restrict__ resid,
             int M, int N, int K,
             long long sA, long long sB, long long sC, long long sR) {
  const int b = blockIdx.z;
  const unsigned short* Ab = A + (size_t)b * sA;
  const unsigned short* Bb = B + (size_t)b * sB;
  const int wave = threadIdx.x >> 5;
  const int lane = threadIdx.x & 31;
  const int col  = lane & 15;
  const int half = lane >> 4;
  const int m0 = (blockIdx.y * 8 + wave) * 16;
  const int n0 = blockIdx.x * 64;
  if (m0 >= M) return;               // uniform per wave

  v8f acc[4];
#pragma unroll
  for (int j = 0; j < 4; ++j) {
    const float bi = bias ? bias[n0 + j * 16 + col] : 0.0f;
#pragma unroll
    for (int v = 0; v < 8; ++v) acc[j][v] = bi;
  }

  const unsigned short* aptr  = Ab + (size_t)(m0 + col) * K + half * 8;
  const unsigned short* bbase = Bb + (size_t)(n0 + col) * K + half * 8;

  // Software-pipelined over k in steps of 64 (two fragment sets in flight).
  Frag16 aA, aB, bA[4], bB[4];
  load_a(aA, aptr);
#pragma unroll
  for (int j = 0; j < 4; ++j) load_b(bA[j], bbase + (size_t)j * 16 * K);

  for (int k0 = 0; k0 < K; k0 += 64) {
    load_a(aB, aptr + k0 + 32);
#pragma unroll
    for (int j = 0; j < 4; ++j)
      load_b(bB[j], bbase + (size_t)j * 16 * K + k0 + 32);
#pragma unroll
    for (int j = 0; j < 4; ++j) acc[j] = wmma_bf16(aA.v, bA[j].v, acc[j]);

    if (k0 + 64 < K) {
      load_a(aA, aptr + k0 + 64);
#pragma unroll
      for (int j = 0; j < 4; ++j)
        load_b(bA[j], bbase + (size_t)j * 16 * K + k0 + 64);
    }
#pragma unroll
    for (int j = 0; j < 4; ++j) acc[j] = wmma_bf16(aB.v, bB[j].v, acc[j]);
  }

  const float alpha = alphaPtr ? *alphaPtr : 1.0f;
  const int mb = m0 + half * 8;       // C frag rows: mb+0..mb+7, col = n
  if (outF) {
    float* o = outF + (size_t)b * sC;
    const float* r = resid ? (resid + (size_t)b * sR) : nullptr;
#pragma unroll
    for (int j = 0; j < 4; ++j) {
      const int n = n0 + j * 16 + col;
#pragma unroll
      for (int v = 0; v < 8; ++v) {
        float val = alpha * acc[j][v];
        if (r) val += r[(size_t)(mb + v) * N + n];
        o[(size_t)(mb + v) * N + n] = val;
      }
    }
  } else {
    unsigned short* o = outH + (size_t)b * sC;
#pragma unroll
    for (int j = 0; j < 4; ++j) {
      const int n = n0 + j * 16 + col;
#pragma unroll
      for (int v = 0; v < 8; ++v)
        o[(size_t)(mb + v) * N + n] = f2bf(acc[j][v]);
    }
  }
}

// ---------------------------------------------------------------------------
// Bilinear: out[t,o] = sum_{i,j} x1[t,i] * W[o,i,j] * x2[t,j]
// Grid: (o = 256, token-chunk = 8). Block = 256 threads (8 waves).
// W[o] (256x256 bf16 = 128 KB) is staged into LDS once per block — via the
// Tensor Data Mover (tensor_load_to_lds + s_wait_tensorcnt) when the builtin
// exists, else a cooperative vector-copy. Each wave then processes 16-token
// tiles: the 8 A fragments (x2 tile) are loaded once and reused across all 16
// i-tiles; B fragments stream from LDS (ds_load_b128). The row-dot with x1
// finishes with a 16-lane xor-shuffle reduction.
// ---------------------------------------------------------------------------
__global__ void __launch_bounds__(256)
bilinear_wmma(const unsigned short* __restrict__ x1,
              const unsigned short* __restrict__ x2,
              const unsigned short* __restrict__ Wb,
              unsigned short* __restrict__ outBf) {
  extern __shared__ unsigned short sW[];           // 256*256 bf16 = 128 KB
  const int o = blockIdx.x;
  const unsigned short* Wo = Wb + (size_t)o * 65536;

#if VMA_USE_TDM
  if (threadIdx.x < 32) {                          // one wave issues the DMA
    const unsigned long long ga = (unsigned long long)(uintptr_t)Wo;
    const unsigned int lds = (unsigned int)(uintptr_t)sW;  // LDS byte offset
    // D# group 0 (ISA 8.3): count=1, lds_addr, 57-bit global addr, type=2.
    u32x4 g0;
    g0.x = 1u;
    g0.y = lds;
    g0.z = (unsigned int)(ga & 0xFFFFFFFFu);
    g0.w = (unsigned int)((ga >> 32) & 0x01FFFFFFu) | (2u << 30);
    // D# group 1 (ISA 8.4): 1-D tensor, data_size=8B, dim0=tile0=16384 elts.
    i32x8 g1;
    g1[0] = (int)(3u << 16);          // [17:16] data_size = 3 (8 bytes)
    g1[1] = (int)(16384u << 16);      // tensor_dim0[15:0]  (bits 63:48)
    g1[2] = (int)(1u << 16);          // tensor_dim1 = 1    (bits 95:80)
    g1[3] = (int)(16384u << 16);      // tile_dim0 = 16384  (bits 127:112)
    g1[4] = 0;                        // tile_dim1/2 unused
    g1[5] = 16384;                    // tensor_dim0_stride[31:0]
    g1[6] = 0;
    g1[7] = 0;
    const i32x4 z4 = {0, 0, 0, 0};
#if __clang_major__ >= 23
    const i32x8 z8 = {0, 0, 0, 0, 0, 0, 0, 0};
    __builtin_amdgcn_tensor_load_to_lds(g0, g1, z4, z4, z8, 0);
#else
    __builtin_amdgcn_tensor_load_to_lds(g0, g1, z4, z4, 0);
#endif
    __builtin_amdgcn_s_wait_tensorcnt(0);
  }
#else
  {
    const u32x4* src = (const u32x4*)Wo;
    u32x4* dst = (u32x4*)sW;
    for (int i = threadIdx.x; i < 8192; i += 256) dst[i] = src[i];
  }
#endif
  __syncthreads();

  const int wave = threadIdx.x >> 5, lane = threadIdx.x & 31;
  const int col = lane & 15, half = lane >> 4;
  const int tokBase = blockIdx.y * 1024;

  for (int tt = wave; tt < 64; tt += 8) {
    const int t0 = tokBase + tt * 16;
    // x2 tile A-fragments: loaded once, reused across all 16 i-tiles.
    const unsigned short* aptr = x2 + (size_t)(t0 + col) * 256 + half * 8;
    Frag16 afr[8];
#pragma unroll
    for (int kk = 0; kk < 8; ++kk) load_a(afr[kk], aptr + kk * 32);

    const unsigned short* x1row = x1 + (size_t)(t0 + half * 8) * 256 + col;
    float part[8];
#pragma unroll
    for (int v = 0; v < 8; ++v) part[v] = 0.f;

    for (int it = 0; it < 16; ++it) {              // i tiles of W[o] rows
      v8f acc = {0.f, 0.f, 0.f, 0.f, 0.f, 0.f, 0.f, 0.f};
      const unsigned short* bp = sW + (size_t)(it * 16 + col) * 256 + half * 8;
#pragma unroll
      for (int kk = 0; kk < 8; ++kk) {             // k = j dimension
        Frag16 bf;
        load_b(bf, bp + kk * 32);
        acc = wmma_bf16(afr[kk].v, bf.v, acc);
      }
      // acc[v] = U[t0+half*8+v, it*16+col]; fold x1 over i.
#pragma unroll
      for (int v = 0; v < 8; ++v)
        part[v] += acc[v] * bf2f(x1row[(size_t)v * 256 + it * 16]);
    }
    // Reduce over the 16 columns within each half (xor masks < 16 keep bit4).
#pragma unroll
    for (int v = 0; v < 8; ++v) {
      float p = part[v];
      p += __shfl_xor(p, 8, 32);
      p += __shfl_xor(p, 4, 32);
      p += __shfl_xor(p, 2, 32);
      p += __shfl_xor(p, 1, 32);
      part[v] = p;
    }
    if (col == 0) {
#pragma unroll
      for (int v = 0; v < 8; ++v)
        outBf[(size_t)(t0 + half * 8 + v) * 256 + o] = f2bf(part[v]);
    }
  }
}

// ---------------------------------------------------------------------------
// Host-side orchestration
// ---------------------------------------------------------------------------
extern "C" void kernel_launch(void* const* d_in, const int* in_sizes, int n_in,
                              void* d_out, int out_size, void* d_ws,
                              size_t ws_size, hipStream_t stream) {
  (void)in_sizes; (void)n_in; (void)out_size; (void)ws_size;
  constexpr int B = 4, S = 2048, D = 256;
  constexpr int TOK = B * S;                     // 8192
  constexpr long long SD  = (long long)S * D;    // 524288  (per-batch [S,D])
  constexpr long long SS  = (long long)S * S;    // 4194304 (per-batch [S,S])
  const float norm = 0.0625f;                    // 1/sqrt(256)

  const float* img   = (const float*)d_in[0];
  const float* audio = (const float*)d_in[1];
  const float* Wqv   = (const float*)d_in[2];
  const float* bqv   = (const float*)d_in[3];
  // d_in[4]=Wqa, d_in[5]=bqa: q_a is unused downstream -> skipped entirely.
  const float* Wkv   = (const float*)d_in[6];
  const float* bkv   = (const float*)d_in[7];
  const float* Wka   = (const float*)d_in[8];
  const float* bka   = (const float*)d_in[9];
  const float* Wbil  = (const float*)d_in[10];
  const float* t_v   = (const float*)d_in[11];
  float* out = (float*)d_out;

  // Workspace sub-allocation (~177 MB total).
  char* w = (char*)d_ws;
  auto take = [&](size_t bytes) -> char* {
    char* p = w; w += (bytes + 255) & ~(size_t)255; return p;
  };
  unsigned short* img_bf  = (unsigned short*)take((size_t)TOK * D * 2);
  unsigned short* aud_bf  = (unsigned short*)take((size_t)TOK * D * 2);
  unsigned short* imgT_bf = (unsigned short*)take((size_t)TOK * D * 2); // [B][D][S]
  unsigned short* audT_bf = (unsigned short*)take((size_t)TOK * D * 2);
  unsigned short* Wqv_bf  = (unsigned short*)take((size_t)D * D * 2);
  unsigned short* Wkv_bf  = (unsigned short*)take((size_t)D * D * 2);
  unsigned short* Wka_bf  = (unsigned short*)take((size_t)D * D * 2);
  unsigned short* Wbil_bf = (unsigned short*)take((size_t)D * D * D * 2); // 33.5 MB
  unsigned short* qv_bf   = (unsigned short*)take((size_t)TOK * D * 2);
  unsigned short* kv_bf   = (unsigned short*)take((size_t)TOK * D * 2);
  unsigned short* ka_bf   = (unsigned short*)take((size_t)TOK * D * 2);
  unsigned short* x1_bf   = (unsigned short*)take((size_t)TOK * D * 2);
  unsigned short* x2_bf   = (unsigned short*)take((size_t)TOK * D * 2);
  unsigned short* aB_bf   = (unsigned short*)take((size_t)TOK * D * 2);
  unsigned short* P_bf    = (unsigned short*)take((size_t)B * SS * 2);   // 33.5 MB
  float*          Sc      = (float*)take((size_t)B * SS * 4);            // 67 MB

  // 1) f32 -> bf16 casts
  cvt_f32_bf16<<<8192, 256, 0, stream>>>(img,  img_bf, (long long)TOK * D);
  cvt_f32_bf16<<<8192, 256, 0, stream>>>(audio, aud_bf, (long long)TOK * D);
  cvt_f32_bf16<<<256,  256, 0, stream>>>(Wqv, Wqv_bf, (long long)D * D);
  cvt_f32_bf16<<<256,  256, 0, stream>>>(Wkv, Wkv_bf, (long long)D * D);
  cvt_f32_bf16<<<256,  256, 0, stream>>>(Wka, Wka_bf, (long long)D * D);
  cvt_f32_bf16<<<65536, 256, 0, stream>>>(Wbil, Wbil_bf, (long long)D * D * D);

  // 2) per-batch transposes so P@V becomes the same NT GEMM form
  transpose_bf16<<<dim3(D / 32, S / 32, B), 256, 0, stream>>>(img_bf, imgT_bf, S, D);
  transpose_bf16<<<dim3(D / 32, S / 32, B), 256, 0, stream>>>(aud_bf, audT_bf, S, D);

  // 3) projections: q_v, k_v, k_a   (M=8192, N=256, K=256)
  gemm_nt_wmma<<<dim3(D / 64, TOK / 128, 1), 256, 0, stream>>>(
      img_bf, Wqv_bf, bqv, nullptr, qv_bf, nullptr, nullptr,
      TOK, D, D, 0, 0, 0, 0);
  gemm_nt_wmma<<<dim3(D / 64, TOK / 128, 1), 256, 0, stream>>>(
      img_bf, Wkv_bf, bkv, nullptr, kv_bf, nullptr, nullptr,
      TOK, D, D, 0, 0, 0, 0);
  gemm_nt_wmma<<<dim3(D / 64, TOK / 128, 1), 256, 0, stream>>>(
      aud_bf, Wka_bf, bka, nullptr, ka_bf, nullptr, nullptr,
      TOK, D, D, 0, 0, 0, 0);

  // 4) atten_va path: S_va = q_v @ k_a^T -> softmax*norm -> x1 = P @ audio
  gemm_nt_wmma<<<dim3(S / 64, S / 128, B), 256, 0, stream>>>(
      qv_bf, ka_bf, nullptr, Sc, nullptr, nullptr, nullptr,
      S, S, D, SD, SD, SS, 0);
  softmax_scale_bf16<<<B * S, 256, 0, stream>>>(Sc, P_bf, S, norm);
  gemm_nt_wmma<<<dim3(D / 64, S / 128, B), 256, 0, stream>>>(
      P_bf, audT_bf, nullptr, nullptr, x1_bf, nullptr, nullptr,
      S, D, S, SS, SD, SD, 0);

  // 5) atten_vv path (reuses Sc/P): x2 = softmax(q_v @ k_v^T)*norm @ img
  gemm_nt_wmma<<<dim3(S / 64, S / 128, B), 256, 0, stream>>>(
      qv_bf, kv_bf, nullptr, Sc, nullptr, nullptr, nullptr,
      S, S, D, SD, SD, SS, 0);
  softmax_scale_bf16<<<B * S, 256, 0, stream>>>(Sc, P_bf, S, norm);
  gemm_nt_wmma<<<dim3(D / 64, S / 128, B), 256, 0, stream>>>(
      P_bf, imgT_bf, nullptr, nullptr, x2_bf, nullptr, nullptr,
      S, D, S, SS, SD, SD, 0);

  // 6) bilinear: atten_B_kv[t,o] = x1_t^T W[o] x2_t  (dominant 275 GFLOP)
  bilinear_wmma<<<dim3(D, 8), 256, 131072, stream>>>(x1_bf, x2_bf, Wbil_bf, aB_bf);

  // 7) third attention with fused epilogue: out = t_v * (P3 @ img) + img
  gemm_nt_wmma<<<dim3(S / 64, S / 128, B), 256, 0, stream>>>(
      aB_bf, img_bf, nullptr, Sc, nullptr, nullptr, nullptr,
      S, S, D, SD, SD, SS, 0);
  softmax_scale_bf16<<<B * S, 256, 0, stream>>>(Sc, P_bf, S, norm);
  gemm_nt_wmma<<<dim3(D / 64, S / 128, B), 256, 0, stream>>>(
      P_bf, imgT_bf, nullptr, out, nullptr, t_v, img,
      S, D, S, SS, SD, SD, SD);
}
